// OLMoE1B7B0924Synapse_9251359555849
// MI455X (gfx1250) — matrix operand order repaired and checked by reading, not compiled, not run on hardware
//
#include <hip/hip_runtime.h>
#include <hip/hip_bf16.h>
#include <math.h>

// ---------------- model constants ----------------
constexpr int kS   = 256;    // tokens
constexpr int kD   = 2048;   // hidden
constexpr int kH   = 16;     // heads
constexpr int kHD  = 128;    // head dim
constexpr int kE   = 64;     // experts
constexpr int kTop = 8;      // top-k
constexpr int kF   = 1024;   // expert intermediate
constexpr float kEps   = 1e-5f;
constexpr float kScale = 0.08838834764831845f;  // 1/sqrt(128)
constexpr float kLn1e4 = 9.210340371976184f;    // ln(10000)

typedef _Float16 half_t;
typedef __attribute__((ext_vector_type(16))) _Float16 v16h;
typedef __attribute__((ext_vector_type(8)))  _Float16 v8h;
typedef __attribute__((ext_vector_type(8)))  float    v8f;
typedef __attribute__((ext_vector_type(4)))  unsigned int v4u;
typedef __attribute__((ext_vector_type(4)))  int      v4i;
typedef __attribute__((ext_vector_type(8)))  int      v8i;

#if defined(__has_builtin)
# if __has_builtin(__builtin_amdgcn_tensor_load_to_lds)
#  define HAVE_TDM 1
# endif
# if __has_builtin(__builtin_amdgcn_s_wait_tensorcnt)
#  define HAVE_TCNT 1
# endif
#endif

// one 16-byte async global->LDS copy (ASYNCcnt-tracked), per lane
__device__ __forceinline__ void async_ld16(unsigned lds_off, const void* gsrc) {
  asm volatile("global_load_async_to_lds_b128 %0, %1, off"
               :: "v"(lds_off), "v"((unsigned long long)(size_t)gsrc)
               : "memory");
}
__device__ __forceinline__ void wait_async0() {
  asm volatile("s_wait_asynccnt 0x0" ::: "memory");
}
__device__ __forceinline__ void wait_tensor0() {
#if defined(HAVE_TCNT)
  __builtin_amdgcn_s_wait_tensorcnt(0);
#else
  asm volatile("s_wait_tensorcnt 0x0" ::: "memory");
#endif
}

// ---------------- reductions ----------------
__device__ __forceinline__ float wave_red_sum(float v) {
  #pragma unroll
  for (int o = 16; o > 0; o >>= 1) v += __shfl_down(v, o, 32);
  return v;
}
__device__ __forceinline__ float wave_red_max(float v) {
  #pragma unroll
  for (int o = 16; o > 0; o >>= 1) v = fmaxf(v, __shfl_down(v, o, 32));
  return v;
}
__device__ float block_sum(float v, float* red) {
  int lane = threadIdx.x & 31, w = threadIdx.x >> 5;
  v = wave_red_sum(v);
  __syncthreads();
  if (lane == 0) red[w] = v;
  __syncthreads();
  float r = (threadIdx.x < (blockDim.x >> 5)) ? red[threadIdx.x] : 0.0f;
  if (w == 0) r = wave_red_sum(r);
  if (threadIdx.x == 0) red[0] = r;
  __syncthreads();
  return red[0];
}
__device__ float block_max(float v, float* red) {
  int lane = threadIdx.x & 31, w = threadIdx.x >> 5;
  v = wave_red_max(v);
  __syncthreads();
  if (lane == 0) red[w] = v;
  __syncthreads();
  float r = (threadIdx.x < (blockDim.x >> 5)) ? red[threadIdx.x] : -INFINITY;
  if (w == 0) r = wave_red_max(r);
  if (threadIdx.x == 0) red[0] = r;
  __syncthreads();
  return red[0];
}

// ---------------- WMMA fragment helpers ----------------
// A fragment (16x32 f16): lane row M = lane&15, K base koff=(lane>>4)*16,
// elements [0..15]=K koff..koff+15.
__device__ __forceinline__ v16h load_a_lds(const half_t* ap) {
  v8h lo = *(const v8h*)(ap);
  v8h hi = *(const v8h*)(ap + 8);
  v16h a;
  #pragma unroll
  for (int i = 0; i < 8; ++i) { a[i] = lo[i]; a[8 + i] = hi[i]; }
  return a;
}
// B fragment (32x16 f16): lane holds 16 consecutive K of column N=lane&15,
// K base (lane>>4)*16.  Source fp32 row-major [N][K]; convert in regs.
__device__ __forceinline__ v16h load_b_f32(const float* p) {
  v16h b;
  #pragma unroll
  for (int i = 0; i < 4; ++i) {
    float4 f = ((const float4*)p)[i];
    b[4*i + 0] = (half_t)f.x;  b[4*i + 1] = (half_t)f.y;
    b[4*i + 2] = (half_t)f.z;  b[4*i + 3] = (half_t)f.w;
  }
  return b;
}

// ---------------- kernel 1: RMSNorm(x) -> f16 ----------------
__global__ void k_rmsnorm_f16(const float* __restrict__ x, const float* __restrict__ w,
                              half_t* __restrict__ out) {
  __shared__ float red[8];
  int t = blockIdx.x;
  const float* xr = x + (size_t)t * kD + threadIdx.x * 8;
  float4 v0 = *(const float4*)(xr);
  float4 v1 = *(const float4*)(xr + 4);
  float ss = v0.x*v0.x + v0.y*v0.y + v0.z*v0.z + v0.w*v0.w
           + v1.x*v1.x + v1.y*v1.y + v1.z*v1.z + v1.w*v1.w;
  ss = block_sum(ss, red);
  float rn = rsqrtf(ss / (float)kD + kEps);
  const float* wr = w + threadIdx.x * 8;
  half_t* o = out + (size_t)t * kD + threadIdx.x * 8;
  float vv[8] = {v0.x, v0.y, v0.z, v0.w, v1.x, v1.y, v1.z, v1.w};
  #pragma unroll
  for (int i = 0; i < 8; ++i) o[i] = (half_t)(vv[i] * rn * wr[i]);
}

// ---------------- kernel 2: C[M,N] = A_f16[M,K] x W_f32[N,K]^T via WMMA ----------
// grid = (M/16, N/128), block = 256 (8 waves).  A tile staged to LDS by the
// Tensor Data Mover (2D descriptor: tile 16 x K halfs), else async-LDS copies.
__global__ void k_gemm(const half_t* __restrict__ A, const float* __restrict__ W,
                       float* __restrict__ C, int N, int Kd) {
  extern __shared__ char smem[];
  half_t* As = (half_t*)smem;                 // [16][Kd]
  int m0 = blockIdx.x * 16;
  int wave = threadIdx.x >> 5, lane = threadIdx.x & 31;
  unsigned ldsBase = __builtin_amdgcn_groupstaticsize();  // dynamic LDS origin
#if defined(HAVE_TDM)
  if (threadIdx.x == 0) {     // TDM issue (EXEC-independent, once per block)
    unsigned long long ga = (unsigned long long)(size_t)(A + (size_t)m0 * Kd);
    v4u g0;
    g0.x = 1u;                                            // count=1, user D#
    g0.y = ldsBase;                                       // lds_addr
    g0.z = (unsigned)(ga & 0xffffffffu);                  // global_addr lo
    g0.w = (unsigned)((ga >> 32) & 0x1ffffffu) | (2u << 30);  // hi | type=2
    v8i g1;
    g1[0] = (int)(1u << 16);                              // data_size=2B
    g1[1] = (int)(((unsigned)Kd & 0xffffu) << 16);        // tensor_dim0 lo16
    g1[2] = (int)((((unsigned)Kd >> 16) & 0xffffu) | (16u << 16)); // d0 hi|dim1
    g1[3] = (int)(((unsigned)Kd & 0xffffu) << 16);        // tile_dim0 = Kd
    g1[4] = 16;                                           // tile_dim1 = 16 rows
    g1[5] = Kd;                                           // tensor_dim0_stride
    g1[6] = 0; g1[7] = 0;
    v4i gz = {0, 0, 0, 0};
#if __clang_major__ >= 23
    v8i gz8 = {0, 0, 0, 0, 0, 0, 0, 0};
    __builtin_amdgcn_tensor_load_to_lds(g0, g1, gz, gz, gz8, 0);
#else
    __builtin_amdgcn_tensor_load_to_lds(g0, g1, gz, gz, 0);
#endif
  }
  wait_tensor0();
#else
  {   // async global->LDS staging, 16B per op, ASYNCcnt-tracked
    int r = threadIdx.x >> 4, cb = threadIdx.x & 15;
    int seg = Kd / 16;
    const half_t* src = A + (size_t)(m0 + r) * Kd + cb * seg;
    unsigned doff = ldsBase + (unsigned)((r * Kd + cb * seg) * 2);
    for (int i = 0; i < seg; i += 8) async_ld16(doff + (unsigned)(i * 2), src + i);
    wait_async0();
  }
#endif
  __syncthreads();
  int col = lane & 15, hi = lane >> 4;
  int n0 = blockIdx.y * 128 + wave * 16;
  const float*  wp = W  + (size_t)(n0 + col) * Kd + hi * 16;
  const half_t* ap = As + col * Kd + hi * 16;
  v8f acc = {};
  for (int k0 = 0; k0 < Kd; k0 += 32) {
    __builtin_prefetch(wp + k0 + 256, 0, 1);
    v16h a = load_a_lds(ap + k0);
    v16h b = load_b_f32(wp + k0);
    acc = __builtin_amdgcn_wmma_f32_16x16x32_f16(false, a, false, b,
                                                 (short)0, acc, false, false);
  }
  int mrow = m0 + hi * 8, n = n0 + col;
  #pragma unroll
  for (int j = 0; j < 8; ++j) C[(size_t)(mrow + j) * N + n] = acc[j];
}

// ---------------- kernel 3: qk RMSNorm + RoPE (in place, fp32) ----------------
__global__ void k_qknorm_rope(float* __restrict__ q, float* __restrict__ k,
                              const float* __restrict__ qn_w,
                              const float* __restrict__ kn_w) {
  __shared__ float red[8];
  int t = blockIdx.x;
  float* qr = q + (size_t)t * kD;
  float* kr = k + (size_t)t * kD;
  float ssq = 0.f, ssk = 0.f;
  int base = threadIdx.x * 8;
  #pragma unroll
  for (int i = 0; i < 8; ++i) {
    float a = qr[base + i], b = kr[base + i];
    ssq += a * a; ssk += b * b;
  }
  float rq = rsqrtf(block_sum(ssq, red) / (float)kD + kEps);
  float rk = rsqrtf(block_sum(ssk, red) / (float)kD + kEps);
  #pragma unroll
  for (int pi = 0; pi < 4; ++pi) {
    int p = threadIdx.x * 4 + pi;       // 1024 rotation pairs
    int head = p >> 6, ii = p & 63;
    int b0 = head * kHD + ii, b1 = b0 + 64;
    float inv = __expf(-(float)ii * (kLn1e4 / 64.0f));  // 10000^(-ii/64)
    float sn, cs;
    __sincosf((float)t * inv, &sn, &cs);
    float q0 = qr[b0] * rq * qn_w[b0], q1 = qr[b1] * rq * qn_w[b1];
    qr[b0] = q0 * cs - q1 * sn;  qr[b1] = q1 * cs + q0 * sn;
    float k0v = kr[b0] * rk * kn_w[b0], k1v = kr[b1] * rk * kn_w[b1];
    kr[b0] = k0v * cs - k1v * sn;  kr[b1] = k1v * cs + k0v * sn;
  }
}

// ---------------- kernel 4: causal attention, one block per (q row, head) -----
__global__ void k_attn(const float* __restrict__ q, const float* __restrict__ k,
                       const float* __restrict__ v, half_t* __restrict__ ctx) {
  __shared__ float qrow[kHD];
  __shared__ float prob[kS];
  __shared__ float red[8];
  int s = blockIdx.x, h = blockIdx.y, tid = threadIdx.x;
  if (tid < kHD) qrow[tid] = q[(size_t)s * kD + h * kHD + tid];
  __syncthreads();
  float sc = -INFINITY;
  if (tid <= s) {
    const float* krw = k + (size_t)tid * kD + h * kHD;
    float acc = 0.f;
    for (int j = 0; j < kHD; j += 4) {
      float4 kv = *(const float4*)(krw + j);
      acc += qrow[j] * kv.x + qrow[j+1] * kv.y + qrow[j+2] * kv.z + qrow[j+3] * kv.w;
    }
    sc = acc * kScale;
  }
  float mx = block_max(sc, red);
  float pe = (tid <= s) ? __expf(sc - mx) : 0.0f;
  float sum = block_sum(pe, red);
  prob[tid] = pe;
  __syncthreads();
  if (tid < kHD) {
    const float* vp = v + h * kHD + tid;
    float acc = 0.f;
    for (int kk = 0; kk <= s; ++kk) acc += prob[kk] * vp[(size_t)kk * kD];
    ctx[(size_t)s * kD + h * kHD + tid] = (half_t)(acc / sum);
  }
}

// ---------------- kernel 5: h = x + attn_out; hn = RMSNorm(h)*w -> f16 --------
__global__ void k_addrms2(const float* __restrict__ x, const float* __restrict__ ao,
                          const float* __restrict__ w, float* __restrict__ hout,
                          half_t* __restrict__ hn) {
  __shared__ float red[8];
  int t = blockIdx.x;
  int base = t * kD + threadIdx.x * 8;
  float vals[8]; float ss = 0.f;
  #pragma unroll
  for (int i = 0; i < 8; ++i) {
    float vv = x[base + i] + ao[base + i];
    vals[i] = vv; ss += vv * vv; hout[base + i] = vv;
  }
  float rn = rsqrtf(block_sum(ss, red) / (float)kD + kEps);
  #pragma unroll
  for (int i = 0; i < 8; ++i)
    hn[base + i] = (half_t)(vals[i] * rn * w[threadIdx.x * 8 + i]);
}

// ---------------- kernel 6: router softmax + top-8 scatter --------------------
__global__ void k_zero_cnt(int* c) { c[threadIdx.x] = 0; }

// 64 threads; hn row staged into dynamic LDS with async-to-LDS loads.
__global__ void k_router(const half_t* __restrict__ hn, const float* __restrict__ rw,
                         int* __restrict__ cnt, int* __restrict__ tok,
                         float* __restrict__ tw) {
  extern __shared__ char rsm[];
  half_t* hrow = (half_t*)rsm;            // [kD] f16
  __shared__ float lg[kE];
  __shared__ float pb[kE];
  int t = blockIdx.x;
  unsigned ldsBase = __builtin_amdgcn_groupstaticsize();
  {
    const half_t* src = hn + (size_t)t * kD + threadIdx.x * 32;
    unsigned doff = ldsBase + (unsigned)(threadIdx.x * 64);
    #pragma unroll
    for (int i = 0; i < 32; i += 8) async_ld16(doff + (unsigned)(i * 2), src + i);
    wait_async0();
  }
  __syncthreads();
  {
    const float* w = rw + (size_t)threadIdx.x * kD;
    float acc = 0.f;
    for (int i = 0; i < kD; i += 4)
      acc += (float)hrow[i]   * w[i]   + (float)hrow[i+1] * w[i+1]
           + (float)hrow[i+2] * w[i+2] + (float)hrow[i+3] * w[i+3];
    lg[threadIdx.x] = acc;
  }
  __syncthreads();
  if (threadIdx.x == 0) {
    float mx = -INFINITY;
    for (int e = 0; e < kE; ++e) mx = fmaxf(mx, lg[e]);
    float s = 0.f;
    for (int e = 0; e < kE; ++e) { pb[e] = __expf(lg[e] - mx); s += pb[e]; }
    float inv = 1.0f / s;
    for (int it = 0; it < kTop; ++it) {          // top-8, no renormalization
      int am = 0; float bv = -1.f;
      for (int e = 0; e < kE; ++e) if (pb[e] > bv) { bv = pb[e]; am = e; }
      int slot = atomicAdd(&cnt[am], 1);
      tok[am * kS + slot] = t;
      tw[am * kS + slot] = bv * inv;
      pb[am] = -1.f;
    }
  }
}

// ---------------- kernel 7: fused sparse MoE expert (gate/up/SiLU/down) -------
// grid = (16 token-tiles, 64 experts), block = 256.  Token rows gathered into
// LDS by TDM gather-mode (16 x 16-bit row indices in D# groups 2/3), else by
// per-lane async-LDS copies.  Mid tile lives in LDS.
__global__ void k_moe(const half_t* __restrict__ hn, const float* __restrict__ gate_w,
                      const float* __restrict__ up_w, const float* __restrict__ down_w,
                      const int* __restrict__ cnt, const int* __restrict__ tok,
                      const float* __restrict__ tw, float* __restrict__ out) {
  int e = blockIdx.y, mt = blockIdx.x;
  int ntok = cnt[e];
  if (mt * 16 >= ntok) return;
  extern __shared__ char smem[];
  half_t* As  = (half_t*)smem;            // [16][2048] gathered activations
  half_t* Mid = As + 16 * kD;             // [16][1024] silu(g)*u
  __shared__ int   ttile[16];
  __shared__ float wtile[16];
  int wave = threadIdx.x >> 5, lane = threadIdx.x & 31;
  unsigned ldsBase = __builtin_amdgcn_groupstaticsize();
  if (threadIdx.x < 16) {
    int slot = mt * 16 + threadIdx.x;
    bool valid = slot < ntok;
    ttile[threadIdx.x] = valid ? tok[e * kS + slot] : 0;
    wtile[threadIdx.x] = valid ? tw[e * kS + slot] : 0.0f;
  }
  __syncthreads();
#if defined(HAVE_TDM)
  if (threadIdx.x == 0) {     // TDM gather: 16 rows of hn selected by token ids
    unsigned long long ga = (unsigned long long)(size_t)hn;
    v4u g0;
    g0.x = 1u | (1u << 31);                               // count=1, gather_mode
    g0.y = ldsBase;                                       // lds_addr -> As
    g0.z = (unsigned)(ga & 0xffffffffu);
    g0.w = (unsigned)((ga >> 32) & 0x1ffffffu) | (2u << 30);
    v8i g1;
    g1[0] = (int)(1u << 16);                              // data_size=2B
    g1[1] = (int)(((unsigned)kD & 0xffffu) << 16);        // tensor_dim0 lo16
    g1[2] = (int)(((unsigned)kS) << 16);                  // tensor_dim1 = kS rows
    g1[3] = (int)(((unsigned)kD & 0xffffu) << 16);        // tile_dim0 = kD
    g1[4] = 16;                                           // tile_dim1 = #indices
    g1[5] = kD;                                           // tensor_dim0_stride
    g1[6] = 0; g1[7] = 0;
    v4i g2, g3;                                           // 16-bit row indices
    #pragma unroll
    for (int j = 0; j < 4; ++j) {
      g2[j] = (ttile[2*j]     & 0xffff) | (ttile[2*j + 1] << 16);
      g3[j] = (ttile[2*j + 8] & 0xffff) | (ttile[2*j + 9] << 16);
    }
#if __clang_major__ >= 23
    v8i gz8 = {0, 0, 0, 0, 0, 0, 0, 0};
    __builtin_amdgcn_tensor_load_to_lds(g0, g1, g2, g3, gz8, 0);
#else
    __builtin_amdgcn_tensor_load_to_lds(g0, g1, g2, g3, 0);
#endif
  }
  wait_tensor0();
#else
  {   // async-LDS gather fallback
    int r = threadIdx.x >> 4, cb = threadIdx.x & 15;
    const half_t* src = hn + (size_t)ttile[r] * kD + cb * 128;
    unsigned doff = ldsBase + (unsigned)((r * kD + cb * 128) * 2);
    for (int i = 0; i < 128; i += 8) async_ld16(doff + (unsigned)(i * 2), src + i);
    wait_async0();
  }
#endif
  __syncthreads();
  int col = lane & 15, hi = lane >> 4;
  const half_t* ap = As + col * kD + hi * 16;
  // phase 1: g,u over F=1024 (8 N-tiles per wave); shared A fragment feeds 2 WMMAs
  for (int jt = 0; jt < 8; ++jt) {
    int n0 = (wave * 8 + jt) * 16;
    const float* gp = gate_w + (size_t)e * kF * kD + (size_t)(n0 + col) * kD + hi * 16;
    const float* up = up_w   + (size_t)e * kF * kD + (size_t)(n0 + col) * kD + hi * 16;
    v8f accg = {}, accu = {};
    for (int k0 = 0; k0 < kD; k0 += 32) {
      __builtin_prefetch(gp + k0 + 256, 0, 1);
      __builtin_prefetch(up + k0 + 256, 0, 1);
      v16h a  = load_a_lds(ap + k0);
      v16h bg = load_b_f32(gp + k0);
      v16h bu = load_b_f32(up + k0);
      accg = __builtin_amdgcn_wmma_f32_16x16x32_f16(false, a, false, bg,
                                                    (short)0, accg, false, false);
      accu = __builtin_amdgcn_wmma_f32_16x16x32_f16(false, a, false, bu,
                                                    (short)0, accu, false, false);
    }
    int mb = hi * 8;
    #pragma unroll
    for (int j = 0; j < 8; ++j) {
      float g = accg[j];
      float mid = (g / (1.0f + __expf(-g))) * accu[j];   // SiLU(g)*u
      Mid[(mb + j) * kF + (n0 + col)] = (half_t)mid;
    }
  }
  __syncthreads();
  // phase 2: y = Mid x down_w[e]^T over D=2048 (16 N-tiles per wave), scatter-add
  const half_t* mp = Mid + col * kF + hi * 16;
  for (int jt = 0; jt < 16; ++jt) {
    int n0 = (wave * 16 + jt) * 16;
    const float* dp = down_w + (size_t)e * kD * kF + (size_t)(n0 + col) * kF + hi * 16;
    v8f acc = {};
    for (int k0 = 0; k0 < kF; k0 += 32) {
      __builtin_prefetch(dp + k0 + 256, 0, 1);
      v16h a = load_a_lds(mp + k0);
      v16h b = load_b_f32(dp + k0);
      acc = __builtin_amdgcn_wmma_f32_16x16x32_f16(false, a, false, b,
                                                   (short)0, acc, false, false);
    }
    int mb = hi * 8, n = n0 + col;
    #pragma unroll
    for (int j = 0; j < 8; ++j) {
      float wv = wtile[mb + j];
      if (wv != 0.0f)
        atomicAdd(&out[(size_t)ttile[mb + j] * kD + n], wv * acc[j]);
    }
  }
}

// ---------------- host: launch pipeline on `stream` ---------------------------
extern "C" void kernel_launch(void* const* d_in, const int* in_sizes, int n_in,
                              void* d_out, int out_size, void* d_ws, size_t ws_size,
                              hipStream_t stream) {
  (void)in_sizes; (void)n_in; (void)out_size; (void)ws_size;
  const float* x        = (const float*)d_in[0];
  const float* ln1_w    = (const float*)d_in[1];
  const float* q_w      = (const float*)d_in[2];
  const float* k_w      = (const float*)d_in[3];
  const float* v_w      = (const float*)d_in[4];
  const float* o_w      = (const float*)d_in[5];
  const float* qn_w     = (const float*)d_in[6];
  const float* kn_w     = (const float*)d_in[7];
  const float* ln2_w    = (const float*)d_in[8];
  const float* router_w = (const float*)d_in[9];
  const float* gate_w   = (const float*)d_in[10];
  const float* up_w     = (const float*)d_in[11];
  const float* down_w   = (const float*)d_in[12];
  float* out = (float*)d_out;

  char* wp = (char*)d_ws;
  auto take = [&](size_t n) { char* r = wp; wp += (n + 255) & ~(size_t)255; return r; };
  half_t* xn  = (half_t*)take((size_t)kS * kD * sizeof(half_t));
  float*  q   = (float*) take((size_t)kS * kD * sizeof(float));
  float*  k   = (float*) take((size_t)kS * kD * sizeof(float));
  float*  v   = (float*) take((size_t)kS * kD * sizeof(float));
  half_t* ctx = (half_t*)take((size_t)kS * kD * sizeof(half_t));
  float*  ao  = (float*) take((size_t)kS * kD * sizeof(float));
  half_t* hn  = (half_t*)take((size_t)kS * kD * sizeof(half_t));
  int*    cnt = (int*)   take((size_t)kE * sizeof(int));
  int*    tok = (int*)   take((size_t)kE * kS * sizeof(int));
  float*  tw  = (float*) take((size_t)kE * kS * sizeof(float));

  const size_t asmem = (size_t)16 * kD * sizeof(half_t);             // 64 KB
  const size_t msmem = (size_t)(16 * kD + 16 * kF) * sizeof(half_t); // 96 KB
  const size_t rsmem = (size_t)kD * sizeof(half_t);                  // 4 KB
  dim3 gg(kS / 16, kD / 128);

  k_rmsnorm_f16<<<kS, 256, 0, stream>>>(x, ln1_w, xn);
  k_gemm<<<gg, 256, asmem, stream>>>(xn, q_w, q, kD, kD);
  k_gemm<<<gg, 256, asmem, stream>>>(xn, k_w, k, kD, kD);
  k_gemm<<<gg, 256, asmem, stream>>>(xn, v_w, v, kD, kD);
  k_qknorm_rope<<<kS, 256, 0, stream>>>(q, k, qn_w, kn_w);
  k_attn<<<dim3(kS, kH), 256, 0, stream>>>(q, k, v, ctx);
  k_gemm<<<gg, 256, asmem, stream>>>(ctx, o_w, ao, kD, kD);
  k_addrms2<<<kS, 256, 0, stream>>>(x, ao, ln2_w, out, hn);
  k_zero_cnt<<<1, kE, 0, stream>>>(cnt);
  k_router<<<kS, 64, rsmem, stream>>>(hn, router_w, cnt, tok, tw);
  k_moe<<<dim3(kS / 16, kE), 256, msmem, stream>>>(hn, gate_w, up_w, down_w,
                                                   cnt, tok, tw, out);
}
